// RNNLayer_41652592837041
// MI455X (gfx1250) — matrix operand order repaired and checked by reading, not compiled, not run on hardware
//
#include <hip/hip_runtime.h>
#include <hip/hip_bf16.h>
#include <math.h>

// Problem constants (match reference: T=1024, B=128, I=256, H=512)
#define TT 1024
#define BB 128
#define II 256
#define HH 512
#define NWG 8        // persistent workgroups (must match launch grid)
#define NTHREADS 128 // 4 waves of 32

typedef __attribute__((ext_vector_type(16))) __bf16 v16bf;
typedef __attribute__((ext_vector_type(8)))  float  v8f;
typedef __attribute__((ext_vector_type(4)))  unsigned int u32x4v;
typedef __attribute__((ext_vector_type(8)))  int i32x8v;
typedef __attribute__((ext_vector_type(4)))  int i32x4v;

union ABFrag { v16bf v; u32x4v q[2]; unsigned int u[8]; };

__device__ __forceinline__ unsigned short f2bf(float f) {
  // round-half-up bf16 (<=0.5 ulp bias; below bf16-GEMM noise floor)
  return (unsigned short)((__float_as_uint(f) + 0x8000u) >> 16);
}
// Pack two f32 -> packed bf16x2 in 3 VALU ops (add, add, v_perm_b32).
__device__ __forceinline__ unsigned int pack2(float lo, float hi) {
  unsigned int a = __float_as_uint(lo) + 0x8000u;
  unsigned int b = __float_as_uint(hi) + 0x8000u;
  return __builtin_amdgcn_perm(b, a, 0x07060302u); // {b[31:16], a[31:16]}
}
__device__ __forceinline__ float fast_tanh(float x) {
  // 1 - 2/(1+e^{2x}): exact at +/-inf, no NaN, v_exp_f32 + rcp
  return 1.0f - 2.0f / (1.0f + __expf(2.0f * x));
}

// ---------------------------------------------------------------------------
// Setup: convert weights f32->bf16 into workspace, zero h0, reset barrier.
// Re-run every launch so graph replay is deterministic.
// ---------------------------------------------------------------------------
__global__ void rnn_setup_kernel(const float* __restrict__ Wi,
                                 const float* __restrict__ Wh,
                                 unsigned short* __restrict__ wi16,
                                 unsigned short* __restrict__ wh16,
                                 unsigned short* __restrict__ h0,
                                 unsigned int* __restrict__ counter) {
  int i = blockIdx.x * blockDim.x + threadIdx.x;
  if (i < HH * II) wi16[i] = f2bf(Wi[i]);
  if (i < HH * HH) wh16[i] = f2bf(Wh[i]);
  if (i < BB * HH) h0[i] = 0;          // h starts at zero
  if (i == 0) *counter = 0u;           // grid-barrier counter
}

// ---------------------------------------------------------------------------
// TDM: DMA one timestep of x (128 rows x 256 f32, row-major, contiguous)
// from global memory into LDS. D# per CDNA5 ISA.
// Tracked by TENSORcnt. 2-D tile -> groups 2/3 unused (zero).
// 6-arg builtin form: (u32x4 g0, i32x8 g1, i32x4 g2, i32x4 g3, i32x8, i32)
// ---------------------------------------------------------------------------
__device__ __forceinline__ void tdm_load_x(const float* src, unsigned int lds_off) {
  unsigned long long ga = (unsigned long long)(size_t)src;
  u32x4v g0 = { 1u,                                   // count=1, user D#
                lds_off,                              // lds_addr
                (unsigned int)ga,                     // global_addr[31:0]
                (unsigned int)(ga >> 32) | (2u << 30) // addr[56:32] | type=2
  };
  i32x8v g1;
  g1[0] = (int)(2u << 16);    // workgroup_mask=0, data_size=2 (4 bytes)
  g1[1] = (int)(256u << 16);  // tensor_dim0[15:0]=256 (bits 63:48)
  g1[2] = (int)(128u << 16);  // tensor_dim1[15:0]=128 (bits 95:80)
  g1[3] = (int)(256u << 16);  // tile_dim0=256 (bits 127:112)
  g1[4] = 128;                // tile_dim1=128, tile_dim2=0
  g1[5] = 256;                // tensor_dim0_stride=256
  g1[6] = 0;
  g1[7] = 0;
  i32x4v z4 = {0, 0, 0, 0};
  i32x8v z8 = {0, 0, 0, 0, 0, 0, 0, 0};
  __builtin_amdgcn_tensor_load_to_lds(g0, g1, z4, z4, z8, 0);
}

// ---------------------------------------------------------------------------
// One K-panel sweep over bf16 A (LDS x_t^bf16 or global h_prev):
//   acc[m] += A(rows m*16..m*16+15, K=0..ldk) x W(cols ncol-slab)^T
// Wave32 WMMA operand k order used by this code (contiguous; A and B agree):
//   A 16x32: M=lane&15; per lane one 32B chunk at k = kb+16*hi
//   B 32x16: N=lane&15; per lane one 32B chunk at k = kb+16*hi (W row-major)
// ---------------------------------------------------------------------------
__device__ __forceinline__ void gemm_panel_bf16(
    v8f acc[8], const unsigned short* A,
    const unsigned short* __restrict__ W, int ldk,
    int ncol, int arow, int khiA, int khiB) {
  for (int kt = 0; kt < ldk / 32; ++kt) {
    const int kb = kt * 32;
    ABFrag bf;
    const unsigned short* wp = W + ncol * ldk + kb + khiB;
    bf.q[0] = *(const u32x4v*)(wp);
    bf.q[1] = *(const u32x4v*)(wp + 8);
#pragma unroll
    for (int m = 0; m < 8; ++m) {
      ABFrag af;
      const unsigned short* ap = A + (m * 16 + arow) * ldk + kb + khiA;
      af.q[0] = *(const u32x4v*)(ap);
      af.q[1] = *(const u32x4v*)(ap + 8);
      acc[m] = __builtin_amdgcn_wmma_f32_16x16x32_bf16(
          false, af.v, false, bf.v, (short)0, acc[m], false, false);
    }
  }
}

// Cooperative in-LDS f32 -> bf16 conversion of the TDM-staged x_t tile.
// Done once per WG (not once per wave): 64 iters/thread of b128 load,
// 2x v_perm pack, b64 store.
__device__ __forceinline__ void convert_x_lds(const float* ldsx,
                                              unsigned int* xbf_u, int tid) {
  const float4* s = (const float4*)ldsx;
  for (int i = tid; i < (BB * II) / 4; i += NTHREADS) {
    float4 f = s[i];
    uint2 p;
    p.x = pack2(f.x, f.y);
    p.y = pack2(f.z, f.w);
    *(uint2*)(xbf_u + 2 * i) = p;
  }
}

// ---------------------------------------------------------------------------
// Persistent fused RNN scan. 8 WGs x 4 waves; wave owns 16 H-columns and all
// 128 batch rows (8 C tiles). One device-wide barrier per timestep; x_{t+1}
// is TDM-DMA'd into LDS concurrently with the recurrent GEMM of step t.
// ---------------------------------------------------------------------------
__global__ __launch_bounds__(NTHREADS, 1) void rnn_scan_kernel(
    const float* __restrict__ x, float* __restrict__ out,
    const unsigned short* __restrict__ wi, const unsigned short* __restrict__ wh,
    unsigned short* __restrict__ hb0, unsigned short* __restrict__ hb1,
    unsigned int* __restrict__ counter) {
  // LDS: [0,128K) x_t as f32 (TDM target); [128K,192K) x_t as bf16
  extern __shared__ float ldsx[];
  unsigned int* xbf_u = (unsigned int*)(ldsx + BB * II);
  const unsigned short* xbf = (const unsigned short*)xbf_u;

  const int tid  = threadIdx.x;
  const int lane = tid & 31;
  const int wave = tid >> 5;
  const int arow = lane & 15;            // A row within tile, B/C column
  const bool hig = lane >= 16;
  const int khiA = hig ? 16 : 0;
  const int khiB = hig ? 16 : 0;
  const int col0 = (blockIdx.x * 4 + wave) * 16; // this wave's H-column slab
  const int ncol = col0 + arow;
  const int mrow_off = hig ? 8 : 0;      // C layout: lanes 16-31 hold M+8

  const unsigned int ldsoff = (unsigned int)(size_t)(void*)ldsx;
  const v8f vzero = {0.f, 0.f, 0.f, 0.f, 0.f, 0.f, 0.f, 0.f};
  v8f acc[8];

  // ---- prologue: DMA x_0, convert to bf16, input projection for t=0 ----
  if (wave == 0) tdm_load_x(x, ldsoff);
  __builtin_amdgcn_s_wait_tensorcnt(0);
  __syncthreads();
  convert_x_lds(ldsx, xbf_u, tid);
  __syncthreads();
#pragma unroll
  for (int m = 0; m < 8; ++m) acc[m] = vzero;
  gemm_panel_bf16(acc, xbf, wi, II, ncol, arow, khiA, khiB);

  for (int t = 0; t < TT; ++t) {
    const unsigned short* hprev = (t & 1) ? hb1 : hb0;
    unsigned short* hnext = (t & 1) ? hb0 : hb1;

    // Kick off the x_{t+1} DMA now: the f32 LDS region was fully consumed by
    // convert_x_lds before the previous device-wide barrier, and the DMA
    // overlaps with the serial-critical recurrent GEMM below.
    if (t + 1 < TT && wave == 0)
      tdm_load_x(x + (size_t)(t + 1) * (BB * II), ldsoff);

    // ---- recurrent GEMM: acc += h_prev @ Wh^T (serial critical path) ----
    gemm_panel_bf16(acc, hprev, wh, HH, ncol, arow, khiA, khiB);

    // ---- tanh + write output (f32, NT streaming) and next-h (bf16) ----
    float* out_t = out + (size_t)t * (BB * HH);
#pragma unroll
    for (int m = 0; m < 8; ++m) {
#pragma unroll
      for (int r = 0; r < 8; ++r) {
        const int brow = m * 16 + r + mrow_off;
        const float hv = fast_tanh(acc[m][r]);
        __builtin_nontemporal_store(hv, &out_t[(size_t)brow * HH + ncol]);
        hnext[(size_t)brow * HH + ncol] = f2bf(hv);
        if (t == TT - 1)  // h_final appended after output[T,B,H]
          __builtin_nontemporal_store(
              hv, &out[(size_t)TT * BB * HH + (size_t)brow * HH + ncol]);
      }
    }
    if (t == TT - 1) break;

    // Make h_new visible at device scope (s_wait_storecnt + global_wb).
    __builtin_amdgcn_fence(__ATOMIC_RELEASE, "agent");

    // Warm L2 for the step after next.
    if (t + 2 < TT) {
      const char* pf = (const char*)(x + (size_t)(t + 2) * (BB * II));
      for (int j = tid; j < 512; j += NTHREADS)
        __builtin_prefetch(pf + j * 256, 0, 0);
    }

    // ---- overlapped: convert x_{t+1}, input projection for step t+1 ----
    __builtin_amdgcn_s_wait_tensorcnt(0);  // x_{t+1} landed in LDS
    __syncthreads();
    convert_x_lds(ldsx, xbf_u, tid);
    __syncthreads();
#pragma unroll
    for (int m = 0; m < 8; ++m) acc[m] = vzero;
    gemm_panel_bf16(acc, xbf, wi, II, ncol, arow, khiA, khiB);

    // ---- device-wide barrier: all 8 WGs finished step t ----
    if (tid == 0) {
      atomicAdd(counter, 1u);
      const unsigned int tgt = (unsigned int)(t + 1) * NWG;
      while (__hip_atomic_load(counter, __ATOMIC_RELAXED,
                               __HIP_MEMORY_SCOPE_AGENT) < tgt)
        __builtin_amdgcn_s_sleep(1);
    }
    __syncthreads();
    // Invalidate read-only WGP$ so h_prev loads see fresh L2 data.
    __builtin_amdgcn_fence(__ATOMIC_ACQUIRE, "agent");
  }
}

// ---------------------------------------------------------------------------
// Workspace layout (bytes): [wi16 128K][wh16 512K][hb0 128K][hb1 128K][ctr]
// Total < 1 MB.
// ---------------------------------------------------------------------------
extern "C" void kernel_launch(void* const* d_in, const int* in_sizes, int n_in,
                              void* d_out, int out_size, void* d_ws,
                              size_t ws_size, hipStream_t stream) {
  const float* x  = (const float*)d_in[0];
  const float* Wi = (const float*)d_in[1];
  const float* Wh = (const float*)d_in[2];
  float* out = (float*)d_out;

  char* ws = (char*)d_ws;
  unsigned short* wi16 = (unsigned short*)(ws + 0);
  unsigned short* wh16 = (unsigned short*)(ws + 131072);
  unsigned short* hb0  = (unsigned short*)(ws + 655360);
  unsigned short* hb1  = (unsigned short*)(ws + 786432);
  unsigned int* counter = (unsigned int*)(ws + 917504);

  rnn_setup_kernel<<<(HH * HH + 255) / 256, 256, 0, stream>>>(
      Wi, Wh, wi16, wh16, hb0, counter);

  // LDS: 128 KB f32 staging + 64 KB bf16 = 192 KB per WG (<= 320 KB).
  rnn_scan_kernel<<<NWG, NTHREADS, BB * II * sizeof(float) + BB * II * 2,
                    stream>>>(x, out, wi16, wh16, hb0, hb1, counter);
}
